// TransformerBlock_56461640073329
// MI455X (gfx1250) — matrix operand order repaired and checked by reading, compiled, not run on hardware
//
#include <hip/hip_runtime.h>
#include <math.h>

// ---------------- problem constants ----------------
#define DIM_    1024
#define NH_     16
#define HD_     64
#define HIDDEN_ 2816
#define B_      2
#define T_      2048
#define TOK_    (B_ * T_)
#define EPS_    1e-6f

// ---------------- WMMA types ----------------
typedef __attribute__((ext_vector_type(16))) __bf16 v16bf;
typedef __attribute__((ext_vector_type(8)))  float  v8f;
typedef unsigned int u32x4 __attribute__((ext_vector_type(4)));
typedef int          i32x8 __attribute__((ext_vector_type(8)));
typedef int          i32x4 __attribute__((ext_vector_type(4)));

union FragBF { v16bf bf; unsigned int u[8]; };

__device__ __forceinline__ v8f wmma_bf(v16bf a, v16bf b, v8f c) {
  return __builtin_amdgcn_wmma_f32_16x16x32_bf16(false, a, false, b, (short)0, c,
                                                 false, false);
}

// f32 -> bf16 bits, round-to-nearest-even
__device__ __forceinline__ unsigned short f2bf(float x) {
  union { float f; unsigned int u; } c; c.f = x;
  unsigned int r = c.u + 0x7fffu + ((c.u >> 16) & 1u);
  return (unsigned short)(r >> 16);
}

// ---------------- TDM: 2D tile load Global -> LDS (bf16 elements) -------------
// D# per CDNA5 ISA ch.8: group0 {count=1, lds_addr, global_addr, type=2},
// group1 {data_size=1(2B), tensor_dim0/1, tile_dim0/1, tensor_dim0_stride}.
// Groups 2/3 zero (tensor <= 2D). Tracked by TENSORcnt.
// Toolchain is the 6-arg (clang-23) flavor: (g0, g1, g2, g3, g_ext, cpol).
__device__ __forceinline__ void tdm_load_2d(const unsigned short* gsrc,
                                            unsigned int lds_off_bytes,
                                            unsigned int tensor_d0,
                                            unsigned int tensor_d1,
                                            unsigned int tile_d0,
                                            unsigned int tile_d1,
                                            unsigned long long stride_d0_elems) {
  const unsigned long long ga = (unsigned long long)(size_t)gsrc;  // byte address
  u32x4 g0;
  g0[0] = 0x1u;                                        // count=1, user mode
  g0[1] = lds_off_bytes;                               // lds_addr (bytes)
  g0[2] = (unsigned int)(ga & 0xffffffffu);            // global_addr[31:0]
  g0[3] = (unsigned int)((ga >> 32) & 0x01ffffffu)     // global_addr[56:32]
          | (2u << 30);                                // type = 2 ("image")
  i32x8 g1;
  g1[0] = (int)(1u << 16);                             // data_size=1 -> 2 bytes
  g1[1] = (int)((tensor_d0 & 0xffffu) << 16);          // tensor_dim0[15:0]
  g1[2] = (int)(((tensor_d0 >> 16) & 0xffffu) |
                ((tensor_d1 & 0xffffu) << 16));        // dim0[31:16] | dim1[15:0]
  g1[3] = (int)(((tensor_d1 >> 16) & 0xffffu) |
                ((tile_d0 & 0xffffu) << 16));          // dim1[31:16] | tile_dim0
  g1[4] = (int)(tile_d1 & 0xffffu);                    // tile_dim1 (tile_dim2=0)
  g1[5] = (int)(unsigned int)(stride_d0_elems & 0xffffffffu);  // dim0_stride lo
  g1[6] = (int)((stride_d0_elems >> 32) & 0xffffu);    // dim0_stride hi
  g1[7] = 0;
  i32x4 z4 = {0, 0, 0, 0};
  i32x8 z8 = {0, 0, 0, 0, 0, 0, 0, 0};
  __builtin_amdgcn_tensor_load_to_lds(g0, g1, z4, z4, z8, 0);
}

// ---------------- fragment loaders (ISA 7.12.2 layouts, wave32) ----------------
// A fragment: 16x32 bf16, row-major source, leading dim lda (elements).
__device__ __forceinline__ v16bf load_frag_a(const unsigned short* A, int lda,
                                             int row_base, int k_base) {
  FragBF f;
  const int lane = threadIdx.x & 31;
  const int r = lane & 15, hv = lane >> 4;
  const unsigned int* p0 =
      (const unsigned int*)(A + (size_t)(row_base + r) * lda + k_base + 8 * hv);
  const unsigned int* p1 =
      (const unsigned int*)(A + (size_t)(row_base + r) * lda + k_base + 16 + 8 * hv);
  f.u[0] = p0[0]; f.u[1] = p0[1]; f.u[2] = p0[2]; f.u[3] = p0[3];
  f.u[4] = p1[0]; f.u[5] = p1[1]; f.u[6] = p1[2]; f.u[7] = p1[3];
  return f.bf;
}

// B fragment where the CONTRACTION axis is contiguous in memory:
// Bmat[kk][n] = P[row_base+n][k_base+kk]; VGPR v<4 packs kk=8*hv+2v,+1; v>=4 packs kk=16+8*hv+2(v-4),+1.
__device__ __forceinline__ v16bf load_frag_kmaj(const unsigned short* P, int ld,
                                                int row_base, int k_base) {
  FragBF f;
  const int lane = threadIdx.x & 31;
  const int n = lane & 15, hv = lane >> 4;
  const unsigned int* p =
      (const unsigned int*)(P + (size_t)(row_base + n) * ld + k_base + 8 * hv);
  const unsigned int* q =
      (const unsigned int*)(P + (size_t)(row_base + n) * ld + k_base + 16 + 8 * hv);
  #pragma unroll
  for (int v = 0; v < 4; ++v) { f.u[v] = p[v]; f.u[v + 4] = q[v]; }
  return f.bf;
}

// ---------------- elementwise kernels ----------------
__global__ void cast_f32_bf16_k(const float* __restrict__ in,
                                unsigned short* __restrict__ out, long n) {
  long i = (long)blockIdx.x * blockDim.x + threadIdx.x;
  if (i < n) out[i] = f2bf(in[i]);
}

// cast f32 [R,C] -> bf16 transposed [C,R] through LDS (32x32 tiles, 256 thr)
__global__ __launch_bounds__(256) void cast_transpose_k(const float* __restrict__ in,
                                                        unsigned short* __restrict__ out,
                                                        int R, int Ccols) {
  __shared__ unsigned short tile[32][33];
  const int c0 = blockIdx.x * 32, r0 = blockIdx.y * 32;
  const int tx = threadIdx.x & 31, ty = threadIdx.x >> 5;
  #pragma unroll
  for (int i = ty; i < 32; i += 8)
    tile[i][tx] = f2bf(in[(size_t)(r0 + i) * Ccols + c0 + tx]);
  __syncthreads();
  #pragma unroll
  for (int i = ty; i < 32; i += 8)
    out[(size_t)(c0 + i) * R + r0 + tx] = tile[tx][i];
}

__global__ void add_k(const float* __restrict__ a, const float* __restrict__ b,
                      float* __restrict__ out, long n) {
  long i = (long)blockIdx.x * blockDim.x + threadIdx.x;
  if (i < n) out[i] = a[i] + b[i];
}

// rmsnorm per token (reference: g * x / (sqrt(sum x^2) + eps)), bf16 output
__global__ __launch_bounds__(256) void rmsnorm_k(const float* __restrict__ x,
                                                 const float* __restrict__ g,
                                                 unsigned short* __restrict__ out) {
  const int t = blockIdx.x;
  const float* xr = x + (size_t)t * DIM_;
  float s = 0.f;
  for (int i = threadIdx.x; i < DIM_; i += 256) { float v = xr[i]; s += v * v; }
  #pragma unroll
  for (int off = 16; off >= 1; off >>= 1) s += __shfl_xor(s, off, 32);
  __shared__ float red[8];
  const int wid = threadIdx.x >> 5;
  if ((threadIdx.x & 31) == 0) red[wid] = s;
  __syncthreads();
  float tot = 0.f;
  #pragma unroll
  for (int w = 0; w < 8; ++w) tot += red[w];
  const float inv = 1.f / (sqrtf(tot) + EPS_);
  for (int i = threadIdx.x; i < DIM_; i += 256)
    out[(size_t)t * DIM_ + i] = f2bf(g[i] * xr[i] * inv);
}

// swiglu: up[:, :H] = proj, up[:, H:] = gate ; act = silu(gate)*proj (bf16)
__global__ __launch_bounds__(256) void swiglu_k(const float* __restrict__ up,
                                                unsigned short* __restrict__ act) {
  const int t = blockIdx.x;
  const float* ur = up + (size_t)t * (2 * HIDDEN_);
  for (int i = threadIdx.x; i < HIDDEN_; i += 256) {
    float pr = ur[i];
    float gt = ur[HIDDEN_ + i];
    float si = gt / (1.f + __expf(-gt));
    act[(size_t)t * HIDDEN_ + i] = f2bf(si * pr);
  }
}

// rope: qkv f32 [TOK, 3*DIM] -> q,k rotary bf16 [B,H,T,HD]; v bf16 TRANSPOSED [B,H,HD,T]
__device__ __forceinline__ float invfreq(int k) {
  return __powf(10000.f, -(float)k * (1.f / 32.f));
}

__global__ __launch_bounds__(256) void rope_k(const float* __restrict__ qkv,
                                              unsigned short* __restrict__ qh,
                                              unsigned short* __restrict__ kh,
                                              unsigned short* __restrict__ vT) {
  const int tok = blockIdx.x;
  const int b = tok >> 11, t = tok & (T_ - 1);
  const float* qr = qkv + (size_t)tok * (3 * DIM_);
  const float* kr = qr + DIM_;
  const float* vr = qr + 2 * DIM_;
  for (int i = threadIdx.x; i < DIM_; i += 256) {
    const int h = i >> 6, d = i & 63, j = d & 31;
    const int second = (d >= 32);
    float sj, cj;
    if (j < 16) { sj = sinf(t * invfreq(2 * j));        cj = sinf(t * invfreq(2 * j + 1)); }
    else { int jj = j - 16; sj = cosf(t * invfreq(2 * jj)); cj = cosf(t * invfreq(2 * jj + 1)); }
    const size_t oidx = (((size_t)(b * NH_ + h)) * T_ + t) * HD_ + d;
    {
      float x1 = qr[h * 64 + 2 * j], x2 = qr[h * 64 + 2 * j + 1];
      qh[oidx] = f2bf(second ? (x1 * sj + x2 * cj) : (x1 * cj - x2 * sj));
    }
    {
      float x1 = kr[h * 64 + 2 * j], x2 = kr[h * 64 + 2 * j + 1];
      kh[oidx] = f2bf(second ? (x1 * sj + x2 * cj) : (x1 * cj - x2 * sj));
    }
    vT[(((size_t)(b * NH_ + h)) * HD_ + d) * T_ + t] = f2bf(vr[h * 64 + d]);
  }
}

// ---------------- TDM-staged WMMA GEMM ----------------------------------------
// C[M,N] = A[M,K](bf16) * Bt[N,K](bf16)^T + bias
// 256 threads = 8 waves (2x4 grid), block tile 128x256, wave tile 64x64.
// Per 32-wide k-step: wave0 TDM-loads A(128x32) + B(256x32) tiles into LDS
// (double buffered, TENSORcnt), all waves ds_load fragments, 16 wmma/wave.
__global__ __launch_bounds__(256) void gemm_tdm_k(const unsigned short* __restrict__ A,
                                                  const unsigned short* __restrict__ Bt,
                                                  const float* __restrict__ bias,
                                                  float* __restrict__ C,
                                                  int M, int N, int K) {
  __shared__ __align__(16) unsigned short Abuf[2][128 * 32];
  __shared__ __align__(16) unsigned short Bbuf[2][256 * 32];

  const int nb0 = blockIdx.x * 256;
  const int mb0 = blockIdx.y * 128;
  const int wid = threadIdx.x >> 5;
  const int wm = wid >> 2, wn = wid & 3;
  const int mw = wm * 64, nw = wn * 64;  // wave tile offsets within block tile
  const int lane = threadIdx.x & 31;
  const int n = lane & 15, hv = lane >> 4;

  v8f acc[4][4] = {};

  if (wid == 0) {
    tdm_load_2d(A  + (size_t)mb0 * K, (unsigned int)(size_t)&Abuf[0][0],
                (unsigned)K, (unsigned)M, 32u, 128u, (unsigned long long)K);
    tdm_load_2d(Bt + (size_t)nb0 * K, (unsigned int)(size_t)&Bbuf[0][0],
                (unsigned)K, (unsigned)N, 32u, 256u, (unsigned long long)K);
  }

  const int steps = K / 32;
  for (int s = 0; s < steps; ++s) {
    const int cur = s & 1, nxt = cur ^ 1;
    if (wid == 0) __builtin_amdgcn_s_wait_tensorcnt(0);
    __syncthreads();  // buffers[cur] ready for all; all done reading buffers[nxt]
    if (wid == 0 && s + 1 < steps) {
      const int ks = (s + 1) * 32;
      tdm_load_2d(A  + (size_t)mb0 * K + ks, (unsigned int)(size_t)&Abuf[nxt][0],
                  (unsigned)K, (unsigned)M, 32u, 128u, (unsigned long long)K);
      tdm_load_2d(Bt + (size_t)nb0 * K + ks, (unsigned int)(size_t)&Bbuf[nxt][0],
                  (unsigned)K, (unsigned)N, 32u, 256u, (unsigned long long)K);
    }
    const unsigned short* As = &Abuf[cur][0];
    const unsigned short* Bs = &Bbuf[cur][0];
    v16bf a[4], b[4];
    #pragma unroll
    for (int i = 0; i < 4; ++i) a[i] = load_frag_a(As, 32, mw + 16 * i, 0);
    #pragma unroll
    for (int j = 0; j < 4; ++j) b[j] = load_frag_kmaj(Bs, 32, nw + 16 * j, 0);
    #pragma unroll
    for (int i = 0; i < 4; ++i)
      #pragma unroll
      for (int j = 0; j < 4; ++j)
        acc[i][j] = wmma_bf(a[i], b[j], acc[i][j]);
  }

  const int mb = mb0 + mw, nb = nb0 + nw;
  #pragma unroll
  for (int j = 0; j < 4; ++j) {
    const float bj = bias[nb + 16 * j + n];
    #pragma unroll
    for (int i = 0; i < 4; ++i) {
      #pragma unroll
      for (int v = 0; v < 8; ++v) {
        const int r = mb + 16 * i + v + 8 * hv;
        C[(size_t)r * N + nb + 16 * j + n] = acc[i][j][v] + bj;
      }
    }
  }
}

// ---------------- flash attention (causal), one wave per 16-row q tile ----------------
// Qh,Kh: [B,H,T,HD] bf16 ; Vt: [B,H,HD,T] bf16 ; O: [B,T,DIM] f32
__global__ __launch_bounds__(32) void attn_k(const unsigned short* __restrict__ Qh,
                                             const unsigned short* __restrict__ Kh,
                                             const unsigned short* __restrict__ Vt,
                                             float* __restrict__ O) {
  const int qt = blockIdx.x, h = blockIdx.y, b = blockIdx.z;
  const size_t head_off = ((size_t)(b * NH_ + h)) * T_ * HD_;
  const unsigned short* Q  = Qh + head_off;
  const unsigned short* Km = Kh + head_off;
  const unsigned short* Vh = Vt + head_off;  // [HD, T]
  float* Ob = O + (size_t)b * T_ * DIM_ + h * HD_;

  const int q0 = qt * 16;
  const int lane = threadIdx.x & 31;
  const int n = lane & 15, hv = lane >> 4;

  const v16bf aq0 = load_frag_a(Q, HD_, q0, 0);
  const v16bf aq1 = load_frag_a(Q, HD_, q0, 32);

  v8f o0 = {}, o1 = {}, o2 = {}, o3 = {};
  float m[8], l[8];
  #pragma unroll
  for (int v = 0; v < 8; ++v) { m[v] = -3.0e38f; l[v] = 0.f; }

  __shared__ __align__(16) unsigned short Pl[16 * 32];

  const float scale = 0.125f;  // 1/sqrt(64)
  const int kend = q0 + 16;    // causal limit (exclusive)
  for (int kb = 0; kb < kend; kb += 32) {
    v8f s0 = {}, s1 = {};
    s0 = wmma_bf(aq0, load_frag_kmaj(Km, HD_, kb,      0),  s0);
    s0 = wmma_bf(aq1, load_frag_kmaj(Km, HD_, kb,      32), s0);
    s1 = wmma_bf(aq0, load_frag_kmaj(Km, HD_, kb + 16, 0),  s1);
    s1 = wmma_bf(aq1, load_frag_kmaj(Km, HD_, kb + 16, 32), s1);

    #pragma unroll
    for (int v = 0; v < 8; ++v) {
      const int row = q0 + v + 8 * hv;
      const int c0 = kb + n, c1 = kb + 16 + n;
      float v0 = (c0 <= row) ? s0[v] * scale : -3.0e38f;
      float v1 = (c1 <= row) ? s1[v] * scale : -3.0e38f;
      float mx = fmaxf(v0, v1);
      #pragma unroll
      for (int off = 8; off >= 1; off >>= 1) mx = fmaxf(mx, __shfl_xor(mx, off, 32));
      const float mnew = fmaxf(m[v], mx);
      const float e0 = __expf(v0 - mnew);
      const float e1 = __expf(v1 - mnew);
      float sum = e0 + e1;
      #pragma unroll
      for (int off = 8; off >= 1; off >>= 1) sum += __shfl_xor(sum, off, 32);
      const float alpha = __expf(m[v] - mnew);
      l[v] = l[v] * alpha + sum;
      m[v] = mnew;
      o0[v] *= alpha; o1[v] *= alpha; o2[v] *= alpha; o3[v] *= alpha;
      Pl[(v + 8 * hv) * 32 + n]      = f2bf(e0);
      Pl[(v + 8 * hv) * 32 + 16 + n] = f2bf(e1);
    }
    const v16bf pa = load_frag_a(Pl, 32, 0, 0);
    o0 = wmma_bf(pa, load_frag_kmaj(Vh, T_, 0,  kb), o0);
    o1 = wmma_bf(pa, load_frag_kmaj(Vh, T_, 16, kb), o1);
    o2 = wmma_bf(pa, load_frag_kmaj(Vh, T_, 32, kb), o2);
    o3 = wmma_bf(pa, load_frag_kmaj(Vh, T_, 48, kb), o3);
  }

  #pragma unroll
  for (int v = 0; v < 8; ++v) {
    const int t = q0 + v + 8 * hv;
    const float linv = 1.f / l[v];
    Ob[(size_t)t * DIM_ + 0  + n] = o0[v] * linv;
    Ob[(size_t)t * DIM_ + 16 + n] = o1[v] * linv;
    Ob[(size_t)t * DIM_ + 32 + n] = o2[v] * linv;
    Ob[(size_t)t * DIM_ + 48 + n] = o3[v] * linv;
  }
}

// ---------------- host orchestration ----------------
extern "C" void kernel_launch(void* const* d_in, const int* in_sizes, int n_in,
                              void* d_out, int out_size, void* d_ws, size_t ws_size,
                              hipStream_t stream) {
  const float* x    = (const float*)d_in[0];
  // d_in[1] = mask (causal; applied analytically in attn_k)
  const float* g1   = (const float*)d_in[2];
  const float* g2   = (const float*)d_in[3];
  const float* wqkv = (const float*)d_in[4];
  const float* bqkv = (const float*)d_in[5];
  const float* wo   = (const float*)d_in[6];
  const float* bo   = (const float*)d_in[7];
  const float* w1   = (const float*)d_in[8];
  const float* b1   = (const float*)d_in[9];
  const float* w2   = (const float*)d_in[10];
  const float* b2   = (const float*)d_in[11];
  float* out = (float*)d_out;

  char* ws = (char*)d_ws;
  size_t off = 0;
  auto carve = [&](size_t bytes) -> void* {
    void* p = ws + off;
    off += (bytes + 255) & ~(size_t)255;
    return p;
  };

  unsigned short* h_bf    = (unsigned short*)carve((size_t)TOK_ * DIM_ * 2);
  unsigned short* wqkvT   = (unsigned short*)carve((size_t)DIM_ * 3 * DIM_ * 2);
  unsigned short* woT     = (unsigned short*)carve((size_t)DIM_ * DIM_ * 2);
  unsigned short* w1T     = (unsigned short*)carve((size_t)DIM_ * 2 * HIDDEN_ * 2);
  unsigned short* w2T     = (unsigned short*)carve((size_t)HIDDEN_ * DIM_ * 2);
  float*          qkv     = (float*)carve((size_t)TOK_ * 3 * DIM_ * 4);
  unsigned short* qh      = (unsigned short*)carve((size_t)TOK_ * DIM_ * 2);
  unsigned short* kh      = (unsigned short*)carve((size_t)TOK_ * DIM_ * 2);
  unsigned short* vT      = (unsigned short*)carve((size_t)TOK_ * DIM_ * 2);
  float*          attn_o  = (float*)carve((size_t)TOK_ * DIM_ * 4);
  unsigned short* ao_bf   = (unsigned short*)carve((size_t)TOK_ * DIM_ * 2);
  float*          proj    = (float*)carve((size_t)TOK_ * DIM_ * 4);
  float*          x1      = (float*)carve((size_t)TOK_ * DIM_ * 4);
  unsigned short* h2_bf   = (unsigned short*)carve((size_t)TOK_ * DIM_ * 2);
  float*          up      = (float*)carve((size_t)TOK_ * 2 * HIDDEN_ * 4);
  unsigned short* act_bf  = (unsigned short*)carve((size_t)TOK_ * HIDDEN_ * 2);
  float*          mlp     = (float*)carve((size_t)TOK_ * DIM_ * 4);

  // weight transpose-casts: in [R=K, C=N] f32 -> out [N, K] bf16
  cast_transpose_k<<<dim3(3 * DIM_ / 32, DIM_ / 32), 256, 0, stream>>>(
      wqkv, wqkvT, DIM_, 3 * DIM_);
  cast_transpose_k<<<dim3(DIM_ / 32, DIM_ / 32), 256, 0, stream>>>(
      wo, woT, DIM_, DIM_);
  cast_transpose_k<<<dim3(2 * HIDDEN_ / 32, DIM_ / 32), 256, 0, stream>>>(
      w1, w1T, DIM_, 2 * HIDDEN_);
  cast_transpose_k<<<dim3(DIM_ / 32, HIDDEN_ / 32), 256, 0, stream>>>(
      w2, w2T, HIDDEN_, DIM_);

  // h = rmsnorm(x, g1) -> bf16
  rmsnorm_k<<<TOK_, 256, 0, stream>>>(x, g1, h_bf);

  // qkv = h @ wqkv + bqkv
  gemm_tdm_k<<<dim3(3 * DIM_ / 256, TOK_ / 128), 256, 0, stream>>>(
      h_bf, wqkvT, bqkv, qkv, TOK_, 3 * DIM_, DIM_);

  // rope + head split -> q,k [B,H,T,HD]; v transposed [B,H,HD,T]
  rope_k<<<TOK_, 256, 0, stream>>>(qkv, qh, kh, vT);

  // flash attention -> attn_o f32 [B,T,DIM]
  attn_k<<<dim3(T_ / 16, NH_, B_), 32, 0, stream>>>(qh, kh, vT, attn_o);

  // proj = attn_o @ wo + bo ; x1 = x + proj
  cast_f32_bf16_k<<<dim3((unsigned)(((long)TOK_ * DIM_ + 255) / 256)), 256, 0, stream>>>(
      attn_o, ao_bf, (long)TOK_ * DIM_);
  gemm_tdm_k<<<dim3(DIM_ / 256, TOK_ / 128), 256, 0, stream>>>(
      ao_bf, woT, bo, proj, TOK_, DIM_, DIM_);
  add_k<<<dim3((unsigned)(((long)TOK_ * DIM_ + 255) / 256)), 256, 0, stream>>>(
      x, proj, x1, (long)TOK_ * DIM_);

  // h2 = rmsnorm(x1, g2) ; up = h2 @ w1 + b1 ; act = silu(gate)*proj
  rmsnorm_k<<<TOK_, 256, 0, stream>>>(x1, g2, h2_bf);
  gemm_tdm_k<<<dim3(2 * HIDDEN_ / 256, TOK_ / 128), 256, 0, stream>>>(
      h2_bf, w1T, b1, up, TOK_, 2 * HIDDEN_, DIM_);
  swiglu_k<<<TOK_, 256, 0, stream>>>(up, act_bf);

  // mlp = act @ w2 + b2 ; out = x1 + mlp
  gemm_tdm_k<<<dim3(DIM_ / 256, TOK_ / 128), 256, 0, stream>>>(
      act_bf, w2T, b2, mlp, TOK_, DIM_, HIDDEN_);
  add_k<<<dim3((unsigned)(((long)TOK_ * DIM_ + 255) / 256)), 256, 0, stream>>>(
      x1, mlp, out, (long)TOK_ * DIM_);
}
